// HP2Mv2_51522427683004
// MI455X (gfx1250) — matrix operand order repaired and checked by reading, not compiled, not run on hardware
//
#include <hip/hip_runtime.h>

// MI455X (gfx1250) implementation.
//
// Algebra: the 3-layer per-head projection has no nonlinearity, so we collapse
//   Weff[c] = Wp1[c] @ Wp2[c] @ Wp3[c]  (240x100), beff[c] = (bp1@Wp2+bp2)@Wp3 + bp3
// once per launch (tiny), reducing main compute 75 GF -> 24 GF. The main GEMM is
//   (B*20 x 240) @ (240 x 300)  done in f16 WMMA (16x16x32, f32 accumulate),
// which makes the kernel ~memory-bound (196 MB @ 23.3 TB/s ~ 8.4 us).
// The gather is analytic (i = CUM[j] + v, valid iff v < VN[j]) and fused into
// the GEMM epilogue together with the 0.5*(ox[...,:3]+ox[...,3:]) add.

typedef _Float16 f16;
typedef __attribute__((ext_vector_type(8)))  _Float16 v8h;
typedef __attribute__((ext_vector_type(16))) _Float16 v16h;
typedef __attribute__((ext_vector_type(8)))  float    v8f;

__constant__ int VN_d[20]  = {45,61,43,45,92,34,41,62,44,44,58,42,40,60,41,35,64,28,50,62};
__constant__ int CUM_d[20] = {0,45,106,149,194,286,320,361,423,467,511,569,611,651,711,752,787,851,879,929};
// total verts = 991

// ---------------------------------------------------------------------------
// Kernel 1: T[c][f][h] = sum_m Wp1[c][f][m]*Wp2[c][m][h]   (f<240)
//           T[c][240][h] = sum_m bp1[c][m]*Wp2[c][m][h] + bp2[c][h]  (bias row)
// T shape [3][241][256] f32 in workspace. ~47M MACs, negligible.
// ---------------------------------------------------------------------------
__global__ void prep_T_kernel(const float* __restrict__ Wp1, const float* __restrict__ bp1,
                              const float* __restrict__ Wp2, const float* __restrict__ bp2,
                              float* __restrict__ T)
{
    int e = blockIdx.x * 256 + threadIdx.x;
    const int TOT = 3 * 241 * 256;
    if (e >= TOT) return;
    int c   = e / (241 * 256);
    int rem = e - c * 241 * 256;
    int f   = rem >> 8;
    int h   = rem & 255;
    const float* w2 = Wp2 + c * 256 * 256 + h;   // stride 256 over m
    float a = 0.f;
    if (f < 240) {
        const float* w1 = Wp1 + (c * 240 + f) * 256;
        for (int m = 0; m < 256; ++m) a = fmaf(w1[m], w2[m * 256], a);
    } else {
        const float* b1 = bp1 + c * 256;
        for (int m = 0; m < 256; ++m) a = fmaf(b1[m], w2[m * 256], a);
        a += bp2[c * 256 + h];
    }
    T[e] = a;
}

// ---------------------------------------------------------------------------
// Kernel 2: Bmat (f16, column-major for WMMA-B fragment loads): [320][256]
//   n = c*100 + v (n<300), k<240 : Weff[c][k][v] = sum_h T[c][k][h]*Wp3[c][h][v]
//   k==240                       : beff[c][v] (bias lane; A provides 1.0 at k=240)
//   all padding (n>=300, k>240)  : 0
// ---------------------------------------------------------------------------
__global__ void prep_B_kernel(const float* __restrict__ T, const float* __restrict__ Wp3,
                              const float* __restrict__ bp3, f16* __restrict__ Bmat)
{
    int e = blockIdx.x * 256 + threadIdx.x;   // 320*256 threads exactly
    int n = e >> 8;
    int k = e & 255;
    float a = 0.f;
    if (n < 300 && k <= 240) {
        int c = n / 100;
        int v = n - c * 100;
        const float* t  = T + (c * 241 + k) * 256;
        const float* w3 = Wp3 + c * 256 * 100 + v;   // stride 100 over h
        for (int h = 0; h < 256; ++h) a = fmaf(t[h], w3[h * 100], a);
        if (k == 240) a += bp3[c * 100 + v];
    }
    Bmat[e] = (f16)a;
}

// ---------------------------------------------------------------------------
// Kernel 3: global-info MLP in f32: zbuf[b][0:100] = relu(relu(g@Wg1+bg1)@Wg2+bg2)
// ---------------------------------------------------------------------------
__global__ void z_mlp_kernel(const float* __restrict__ g, const float* __restrict__ Wg1,
                             const float* __restrict__ bg1, const float* __restrict__ Wg2,
                             const float* __restrict__ bg2, float* __restrict__ zbuf)
{
    __shared__ float sg[60];
    __shared__ float sz1[256];
    const int b = blockIdx.x;
    const int t = threadIdx.x;
    if (t < 60) sg[t] = g[b * 60 + t];
    __syncthreads();
    float a = bg1[t];
    #pragma unroll
    for (int i = 0; i < 60; ++i) a = fmaf(sg[i], Wg1[i * 256 + t], a);
    sz1[t] = fmaxf(a, 0.f);
    __syncthreads();
    if (t < 100) {
        float a2 = bg2[t];
        for (int i = 0; i < 256; ++i) a2 = fmaf(sz1[i], Wg2[i * 100 + t], a2);
        zbuf[b * 100 + t] = fmaxf(a2, 0.f);
    }
}

// ---------------------------------------------------------------------------
// Kernel 4: main WMMA GEMM + fused gather epilogue.
//   A: inp rows (x[140] ‖ z[100] ‖ 1.0 ‖ 0-pad) -> K=256, staged in LDS as f16.
//   B: Bmat [320][256] f16 (L2-hot, ~160KB).
//   Block = 256 thr = 8 waves; tile 64 rows x 320 cols; wave = 16 x 160 (10 tiles).
//   8 k-steps of v_wmma_f32_16x16x32_f16 -> 80 WMMA per wave.
// ---------------------------------------------------------------------------
__global__ void __launch_bounds__(256)
main_gemm_kernel(const float* __restrict__ x, const float* __restrict__ zbuf,
                 const float* __restrict__ ox, const f16* __restrict__ Bmat,
                 float* __restrict__ out, int B)
{
    __shared__ f16 smem[64][264];            // 264-stride: conflict-free b128 LDS reads
    const int MROWS = B * 20;
    const int row0  = blockIdx.x * 64;
    const int tid   = threadIdx.x;

    // ---- stage 64 padded inp rows to LDS as f16 (coalesced over k) ----
    for (int idx = tid; idx < 64 * 256; idx += 256) {
        int k = idx & 255;
        int r = idx >> 8;
        int grow = row0 + r;
        float val = 0.f;
        if (grow < MROWS) {
            int b = grow / 20;
            if (k < 140)        val = x[grow * 140 + k];
            else if (k < 240)   val = zbuf[b * 100 + (k - 140)];
            else if (k == 240)  val = 1.0f;      // bias lane
        }
        smem[r][k] = (f16)val;
    }
    __syncthreads();

    const int lane = tid & 31;
    const int wid  = tid >> 5;
    const int rt   = (wid & 3) * 16;     // wave row-tile base within block
    const int n0   = (wid >> 2) * 160;   // wave col base (2 N-halves of 160)
    const int lcol = lane & 15;
    const int hs   = lane >> 4;          // lane half select

    v8f acc[10];
    #pragma unroll
    for (int t = 0; t < 10; ++t) acc[t] = (v8f){0.f,0.f,0.f,0.f,0.f,0.f,0.f,0.f};

    // A fragment: lane holds K[hs*8 + 0..7] and K[16 + hs*8 + 0..7] of row (rt+lcol)
    const f16* arow = &smem[rt + lcol][hs * 8];

    #pragma unroll
    for (int kk = 0; kk < 8; ++kk) {
        v8h alo = *(const v8h*)(arow + kk * 32);
        v8h ahi = *(const v8h*)(arow + kk * 32 + 16);
        v16h afrag = __builtin_shufflevector(alo, ahi, 0,1,2,3,4,5,6,7,8,9,10,11,12,13,14,15);
        #pragma unroll
        for (int t = 0; t < 10; ++t) {
            // B fragment: lane holds K[hs*8 + 0..7] and K[16 + hs*8 + 0..7] of column (n0 + t*16 + lcol)
            const f16* bp = Bmat + (size_t)(n0 + t * 16 + lcol) * 256 + kk * 32 + hs * 8;
            v8h blo = *(const v8h*)(bp);
            v8h bhi = *(const v8h*)(bp + 16);
            v16h bfrag = __builtin_shufflevector(blo, bhi, 0,1,2,3,4,5,6,7,8,9,10,11,12,13,14,15);
            acc[t] = __builtin_amdgcn_wmma_f32_16x16x32_f16(
                false, afrag, false, bfrag, (short)0, acc[t], false, false);
        }
    }

    // ---- fused gather epilogue ----
    // C/D layout: element (vgpr e, lane) <-> row M = e + 8*hs, col N = lcol.
    #pragma unroll
    for (int t = 0; t < 10; ++t) {
        int col = n0 + t * 16 + lcol;          // = c*100 + v
        if (col < 300) {
            int c = col / 100;
            int v = col - c * 100;
            #pragma unroll
            for (int e = 0; e < 8; ++e) {
                int grow = row0 + rt + e + hs * 8;   // = b*20 + j
                if (grow < MROWS) {
                    int b = grow / 20;
                    int j = grow - b * 20;
                    if (v < VN_d[j]) {
                        float pts = 0.5f * (ox[grow * 6 + c] + ox[grow * 6 + 3 + c]);
                        out[(size_t)(b * 991 + CUM_d[j] + v) * 3 + c] = acc[t][e] + pts;
                    }
                }
            }
        }
    }
}

// ---------------------------------------------------------------------------
extern "C" void kernel_launch(void* const* d_in, const int* in_sizes, int n_in,
                              void* d_out, int out_size, void* d_ws, size_t ws_size,
                              hipStream_t stream)
{
    const float* x   = (const float*)d_in[0];
    const float* glo = (const float*)d_in[1];
    const float* ox  = (const float*)d_in[2];
    const float* Wg1 = (const float*)d_in[3];
    const float* bg1 = (const float*)d_in[4];
    const float* Wg2 = (const float*)d_in[5];
    const float* bg2 = (const float*)d_in[6];
    const float* Wp1 = (const float*)d_in[7];
    const float* bp1 = (const float*)d_in[8];
    const float* Wp2 = (const float*)d_in[9];
    const float* bp2 = (const float*)d_in[10];
    const float* Wp3 = (const float*)d_in[11];
    const float* bp3 = (const float*)d_in[12];
    float* out = (float*)d_out;

    const int B = in_sizes[0] / (20 * 140);   // 8192

    // workspace layout (all 16B aligned): zbuf f32 | T f32 | Bmat f16
    char*  ws   = (char*)d_ws;
    float* zbuf = (float*)ws;                                   // B*100 f32
    size_t off  = (size_t)B * 100 * sizeof(float);
    float* T    = (float*)(ws + off);                           // 3*241*256 f32
    off        += (size_t)3 * 241 * 256 * sizeof(float);
    f16*   Bmat = (f16*)(ws + off);                             // 320*256 f16

    prep_T_kernel<<<(3 * 241 * 256 + 255) / 256, 256, 0, stream>>>(Wp1, bp1, Wp2, bp2, T);
    prep_B_kernel<<<320, 256, 0, stream>>>(T, Wp3, bp3, Bmat);
    z_mlp_kernel<<<B, 256, 0, stream>>>(glo, Wg1, bg1, Wg2, bg2, zbuf);

    const int mrows = B * 20;
    main_gemm_kernel<<<(mrows + 63) / 64, 256, 0, stream>>>(x, zbuf, ox, Bmat, out, B);
}
